// dmGCN_10917806866467
// MI455X (gfx1250) — matrix operand order repaired and checked by reading, not compiled, not run on hardware
//
#include <hip/hip_runtime.h>
#include <stddef.h>

// ---------------------------------------------------------------------------
// Problem constants (from reference): N=8192, F=512, H=512, C=64
// ---------------------------------------------------------------------------
#define NN      8192
#define FF      512
#define HH      512
#define CC      64
#define MAXNNZ  256
// k-th smallest (1-indexed) over |S0|,|S1|,|S2| concat: n=557056,
// k = 1 + round(0.5*(n-1)) = 1 + 278528 (banker's rounding of 278527.5)
#define SEL_K   278529u

#if __has_builtin(__builtin_amdgcn_tensor_load_to_lds) && \
    __has_builtin(__builtin_amdgcn_s_wait_tensorcnt)
#define USE_TDM 1
#else
#define USE_TDM 0
#endif

typedef __attribute__((ext_vector_type(16))) __bf16 bf16x16;
typedef __attribute__((ext_vector_type(8)))  __bf16 bf16x8;
typedef __attribute__((ext_vector_type(8)))  float  f32x8;
typedef __attribute__((ext_vector_type(4)))  unsigned int u32x4;
typedef __attribute__((ext_vector_type(8)))  int i32x8;
typedef __attribute__((ext_vector_type(4)))  int i32x4;

__device__ __forceinline__ unsigned short f2bf(float f) {
    unsigned int u = __float_as_uint(f);
    unsigned int r = u + 0x7fffu + ((u >> 16) & 1u);   // round-to-nearest-even
    return (unsigned short)(r >> 16);
}

// ---------------------------------------------------------------------------
// Utility zero-fill kernels (graph-capture safe; no hipMemsetAsync needed)
// ---------------------------------------------------------------------------
__global__ void zero_u32(unsigned int* p, int n) {
    int i = blockIdx.x * blockDim.x + threadIdx.x;
    if (i < n) p[i] = 0u;
}
__global__ void zero_f32(float* p, int n) {
    int i = blockIdx.x * blockDim.x + threadIdx.x;
    if (i < n) p[i] = 0.0f;
}

// ---------------------------------------------------------------------------
// Exact global median of |S| via 2-level radix select on float bit patterns
// (|s| >= 0, so uint order == float order)
// ---------------------------------------------------------------------------
__global__ void hist_hi(const float* __restrict__ S, int n, unsigned int* __restrict__ hist) {
    int i = blockIdx.x * blockDim.x + threadIdx.x;
    if (i < n) {
        unsigned int key = __float_as_uint(S[i]) & 0x7fffffffu;
        atomicAdd(&hist[key >> 16], 1u);
    }
}
__global__ void sel_hi(const unsigned int* __restrict__ hist, unsigned int k,
                       unsigned int* __restrict__ sel) {
    if (threadIdx.x == 0 && blockIdx.x == 0) {
        unsigned int cum = 0;
        for (int b = 0; b < 65536; ++b) {
            unsigned int c = hist[b];
            if (cum + c >= k) { sel[0] = (unsigned int)b; sel[1] = k - cum; return; }
            cum += c;
        }
        sel[0] = 65535u; sel[1] = 1u;
    }
}
__global__ void hist_lo(const float* __restrict__ S, int n,
                        const unsigned int* __restrict__ sel,
                        unsigned int* __restrict__ hist2) {
    int i = blockIdx.x * blockDim.x + threadIdx.x;
    if (i < n) {
        unsigned int key = __float_as_uint(S[i]) & 0x7fffffffu;
        if ((key >> 16) == sel[0]) atomicAdd(&hist2[key & 0xffffu], 1u);
    }
}
__global__ void sel_lo(const unsigned int* __restrict__ hist2,
                       const unsigned int* __restrict__ sel,
                       float* __restrict__ thr) {
    if (threadIdx.x == 0 && blockIdx.x == 0) {
        unsigned int k = sel[1], cum = 0;
        for (int b = 0; b < 65536; ++b) {
            unsigned int c = hist2[b];
            if (cum + c >= k) { *thr = __uint_as_float((sel[0] << 16) | (unsigned int)b); return; }
            cum += c;
        }
        *thr = __uint_as_float((sel[0] << 16) | 0xffffu);
    }
}

// ---------------------------------------------------------------------------
// Wm = W * (|S| > thr), stored bf16 row-major [O,K]; plus f32->bf16 convert
// ---------------------------------------------------------------------------
__global__ void mask_w(const float* __restrict__ W, const float* __restrict__ S,
                       const float* __restrict__ thr, unsigned short* __restrict__ Wm, int n) {
    int i = blockIdx.x * blockDim.x + threadIdx.x;
    if (i < n) {
        float t = *thr;
        float w = (fabsf(S[i]) > t) ? W[i] : 0.0f;
        Wm[i] = f2bf(w);
    }
}
__global__ void cvt_bf(const float* __restrict__ X, unsigned short* __restrict__ Xb, int n) {
    int i = blockIdx.x * blockDim.x + threadIdx.x;
    if (i < n) Xb[i] = f2bf(X[i]);
}

// ---------------------------------------------------------------------------
// bf16 WMMA GEMM:  C[M,N] = A[M,K] * Wm[N,K]^T   (both bf16 row-major)
// - A panel (block_rows x 32) staged into LDS by the Tensor Data Mover
//   (wave 0 issues tensor_load_to_lds, TENSORcnt-tracked), consumed with
//   ds_load_b128 per the 16-bit A 16x32 VGPR layout.
// - B fragments streamed with global_load_b128 (weights live in L2).
// - Wave tile: 2(M) x 4(N) accumulators of v_wmma_f32_16x16x32_bf16.
// ---------------------------------------------------------------------------
template<int WM_W, int WN_W>
__global__ __launch_bounds__(WM_W * WN_W * 32)
void gemm_bf16(const unsigned short* __restrict__ Au,
               const unsigned short* __restrict__ Bu,
               float* __restrict__ Cf, unsigned short* __restrict__ Cbf,
               int M, int N, int K) {
    constexpr int BLK_ROWS = WM_W * 32;
    const __bf16* Bt = reinterpret_cast<const __bf16*>(Bu);
    const int lane = threadIdx.x & 31;
    const int wave = threadIdx.x >> 5;
    const int wm = wave / WN_W, wn = wave % WN_W;
    const long rowB = (long)blockIdx.y * BLK_ROWS;          // block row panel
    const long row0 = rowB + wm * 32;
    const long col0 = (long)blockIdx.x * (WN_W * 64) + wn * 64;
    const int  lr = lane & 15;
    const bool hi = lane >= 16;

#if USE_TDM
    __shared__ __align__(16) unsigned short sA[BLK_ROWS * 32];   // A panel tile
    const __bf16* sAp = reinterpret_cast<const __bf16*>(&sA[0]);
#else
    const __bf16* A = reinterpret_cast<const __bf16*>(Au);
#endif

    f32x8 acc[2][4];
#pragma unroll
    for (int i = 0; i < 2; ++i)
#pragma unroll
        for (int j = 0; j < 4; ++j)
#pragma unroll
            for (int e = 0; e < 8; ++e) acc[i][j][e] = 0.0f;

    for (int k0 = 0; k0 < K; k0 += 32) {
#if USE_TDM
        __syncthreads();                 // prior consumers done before overwrite
        if (wave == 0) {
            // ---- Tensor DMA descriptor (D#), 2-D tile: 32(K) x BLK_ROWS ----
            unsigned long long gaddr = (unsigned long long)(uintptr_t)Au +
                ((unsigned long long)rowB * (unsigned long long)K + (unsigned long long)k0) * 2ull;
            unsigned int ldsoff = (unsigned int)(uintptr_t)(void*)&sA[0];
            u32x4 g0;
            g0[0] = 1u;                                            // count=1 (valid)
            g0[1] = ldsoff;                                        // lds_addr
            g0[2] = (unsigned int)(gaddr & 0xffffffffull);         // global_addr lo
            g0[3] = (unsigned int)((gaddr >> 32) & 0x01ffffffull)  // global_addr hi
                    | 0x80000000u;                                 // type=2 (image)
            i32x8 g1;
            g1[0] = 0x00010000;                                    // data_size=2B
            g1[1] = (int)(((unsigned)K & 0xffffu) << 16);          // tensor_dim0 lo16
            g1[2] = (int)((((unsigned)K >> 16) & 0xffffu) |
                          (((unsigned)M & 0xffffu) << 16));        // dim0 hi | dim1 lo
            g1[3] = (int)((((unsigned)M >> 16) & 0xffffu) |
                          (32u << 16));                            // dim1 hi | tile_dim0=32
            g1[4] = (int)BLK_ROWS;                                 // tile_dim1 (tile_dim2=0)
            g1[5] = (int)K;                                        // tensor_dim0_stride lo
            g1[6] = 0;                                             // stride hi / dim1_stride
            g1[7] = 0;
            i32x4 gz4 = {0, 0, 0, 0};                              // 2-D: groups 2/3 unused
            i32x8 gz8 = {0, 0, 0, 0, 0, 0, 0, 0};
            __builtin_amdgcn_tensor_load_to_lds(g0, g1, gz4, gz4, gz8, 0);
            __builtin_amdgcn_s_wait_tensorcnt(0);
        }
        __syncthreads();                 // panel visible to all waves
#endif
        // A fragments (16-bit A 16x32 layout): lanes 0-15 carry K k0+[0..7],[16..23];
        // lanes 16-31 carry K k0+[8..15],[24..31]  -> two contiguous 16B loads.
        bf16x16 afrag[2];
#pragma unroll
        for (int mt = 0; mt < 2; ++mt) {
#if USE_TDM
            long prow = wm * 32 + mt * 16 + lr;                    // row in LDS panel
            long base = prow * 32 + (hi ? 8 : 0);
            bf16x8 lo8 = *reinterpret_cast<const bf16x8*>(sAp + base);
            bf16x8 hi8 = *reinterpret_cast<const bf16x8*>(sAp + base + 16);
#else
            long r = row0 + mt * 16 + lr;
            long base = r * (long)K + k0 + (hi ? 8 : 0);
            bf16x8 lo8 = *reinterpret_cast<const bf16x8*>(A + base);
            bf16x8 hi8 = *reinterpret_cast<const bf16x8*>(A + base + 16);
#endif
#pragma unroll
            for (int t = 0; t < 8; ++t) { afrag[mt][t] = lo8[t]; afrag[mt][8 + t] = hi8[t]; }
        }
        // B fragments (32x16 B layout): lane n = lr, K = k0 + (hi?8:0) + [0..7],[16..23]
        // -> two 16B loads of a Wm row.
        bf16x16 bfrag[4];
#pragma unroll
        for (int nt = 0; nt < 4; ++nt) {
            long c = col0 + nt * 16 + lr;
            long base = c * (long)K + k0 + (hi ? 8 : 0);
            bf16x8 lo8 = *reinterpret_cast<const bf16x8*>(Bt + base);
            bf16x8 hi8 = *reinterpret_cast<const bf16x8*>(Bt + base + 16);
#pragma unroll
            for (int t = 0; t < 8; ++t) { bfrag[nt][t] = lo8[t]; bfrag[nt][8 + t] = hi8[t]; }
        }
        if (k0 + 32 < K)
            __builtin_prefetch(Bt + (col0 + lr) * (long)K + k0 + 32, 0, 1);
#pragma unroll
        for (int mt = 0; mt < 2; ++mt)
#pragma unroll
            for (int nt = 0; nt < 4; ++nt)
                acc[mt][nt] = __builtin_amdgcn_wmma_f32_16x16x32_bf16(
                    false, afrag[mt], false, bfrag[nt],
                    (short)0, acc[mt][nt], false, false);
    }

    // C/D layout: VGPR e holds M = e (+8 for upper half-wave), N = lane%16.
#pragma unroll
    for (int mt = 0; mt < 2; ++mt)
#pragma unroll
        for (int nt = 0; nt < 4; ++nt) {
            long col = col0 + nt * 16 + lr;
#pragma unroll
            for (int e = 0; e < 8; ++e) {
                long row = row0 + mt * 16 + e + (hi ? 8 : 0);
                float v = acc[mt][nt][e];
                if (Cf)  Cf[row * (long)N + col] = v;
                if (Cbf) Cbf[row * (long)N + col] = f2bf(v);
            }
        }
}

// ---------------------------------------------------------------------------
// One pass over dense adj (268 MB, read exactly once): compact nonzero column
// indices per row (self-loop forced) + degree (= count, adj is binary)
// ---------------------------------------------------------------------------
__global__ void scan_adj(const float* __restrict__ adj, int* __restrict__ idxs,
                         int* __restrict__ cnt, float* __restrict__ dinv, int n) {
    int i = blockIdx.x;
    __shared__ int c;
    if (threadIdx.x == 0) c = 0;
    __syncthreads();
    for (int j = threadIdx.x; j < n; j += blockDim.x) {
        float v = adj[(size_t)i * n + j];
        bool nz = (v != 0.0f) || (j == i);      // diag forced to 1
        if (nz) {
            int p = atomicAdd(&c, 1);
            if (p < MAXNNZ) idxs[(size_t)i * MAXNNZ + p] = j;
        }
    }
    __syncthreads();
    if (threadIdx.x == 0) {
        int cc = c > MAXNNZ ? MAXNNZ : c;
        cnt[i] = cc;
        float d = (float)(c < 1 ? 1 : c);
        dinv[i] = rsqrtf(d);
    }
}

__global__ void make_wts(const int* __restrict__ idxs, const int* __restrict__ cnt,
                         const float* __restrict__ dinv, float* __restrict__ wts) {
    int i = blockIdx.x;
    float di = dinv[i];
    int c = cnt[i];
    for (int p = threadIdx.x; p < c; p += blockDim.x)
        wts[(size_t)i * MAXNNZ + p] = di * dinv[idxs[(size_t)i * MAXNNZ + p]];
}

// ---------------------------------------------------------------------------
// SpMM: out[i,:] = sum_p w[i,p] * H[idx[i,p], :]   (gathers hit L2; H << 192MB)
// ---------------------------------------------------------------------------
__global__ void spmm(const int* __restrict__ idxs, const float* __restrict__ wts,
                     const int* __restrict__ cnt, const float* __restrict__ Hm,
                     float* __restrict__ out, int HC) {
    int i = blockIdx.x;
    int c = cnt[i];
    __shared__ int   sj[MAXNNZ];
    __shared__ float sw[MAXNNZ];
    for (int p = threadIdx.x; p < c; p += blockDim.x) {
        sj[p] = idxs[(size_t)i * MAXNNZ + p];
        sw[p] = wts[(size_t)i * MAXNNZ + p];
    }
    __syncthreads();
    for (int col = threadIdx.x; col < HC; col += blockDim.x) {
        float acc = 0.0f;
        for (int p = 0; p < c; ++p)
            acc += sw[p] * Hm[(size_t)sj[p] * HC + col];
        out[(size_t)i * HC + col] = acc;
    }
}

// ---------------------------------------------------------------------------
// PairNorm pieces
// ---------------------------------------------------------------------------
__global__ void col_sum(const float* __restrict__ A, float* __restrict__ colsum,
                        int rows_per_blk, int HC) {
    size_t r0 = (size_t)blockIdx.x * rows_per_blk;
    for (int col = threadIdx.x; col < HC; col += blockDim.x) {
        float s = 0.0f;
        for (int r = 0; r < rows_per_blk; ++r) s += A[(r0 + r) * HC + col];
        atomicAdd(&colsum[col], s);
    }
}
__global__ void sq_sum(const float* __restrict__ A, const float* __restrict__ colsum,
                       float* __restrict__ tot, int rows_per_blk, int HC, float invN) {
    __shared__ float red[256];
    size_t base = (size_t)blockIdx.x * rows_per_blk * HC;
    int total = rows_per_blk * HC;
    float s = 0.0f;
    for (int t = threadIdx.x; t < total; t += blockDim.x) {
        int col = t % HC;
        float v = A[base + t] - colsum[col] * invN;
        s += v * v;
    }
    red[threadIdx.x] = s;
    __syncthreads();
    for (int st = blockDim.x / 2; st > 0; st >>= 1) {
        if ((int)threadIdx.x < st) red[threadIdx.x] += red[threadIdx.x + st];
        __syncthreads();
    }
    if (threadIdx.x == 0) atomicAdd(tot, red[0]);
}
// h2 = bf16(relu((agg - colmean) / sqrt(1e-6 + mean rownorm^2)))
__global__ void pn_apply(const float* __restrict__ agg, const float* __restrict__ colsum,
                         const float* __restrict__ tot, unsigned short* __restrict__ h2,
                         int total, int HC, float invN) {
    int i = blockIdx.x * blockDim.x + threadIdx.x;
    if (i < total) {
        int col = i % HC;
        float mean  = colsum[col] * invN;
        float scale = rsqrtf(1e-6f + (*tot) * invN);
        float v = (agg[i] - mean) * scale;
        v = v > 0.0f ? v : 0.0f;
        h2[i] = f2bf(v);
    }
}

// ---------------------------------------------------------------------------
// Host orchestration
// ---------------------------------------------------------------------------
extern "C" void kernel_launch(void* const* d_in, const int* in_sizes, int n_in,
                              void* d_out, int out_size, void* d_ws, size_t ws_size,
                              hipStream_t stream) {
    const float* x   = (const float*)d_in[0];
    const float* adj = (const float*)d_in[1];
    const float* W0  = (const float*)d_in[2];
    const float* W1  = (const float*)d_in[3];
    const float* W2  = (const float*)d_in[4];
    const float* S0  = (const float*)d_in[5];
    const float* S1  = (const float*)d_in[6];
    const float* S2  = (const float*)d_in[7];
    float* out = (float*)d_out;   // [N, C] f32

    // workspace carve-out
    char* ws = (char*)d_ws;
    size_t off = 0;
    auto carve = [&](size_t bytes) -> void* {
        void* p = ws + off;
        off = (off + bytes + 255) & ~(size_t)255;
        return p;
    };
    unsigned int* hist  = (unsigned int*)carve(65536 * sizeof(unsigned int));
    unsigned int* hist2 = (unsigned int*)carve(65536 * sizeof(unsigned int));
    unsigned int* sel   = (unsigned int*)carve(2 * sizeof(unsigned int));
    float* thr          = (float*)carve(sizeof(float));
    unsigned short* xbf = (unsigned short*)carve((size_t)NN * FF * 2);
    unsigned short* w0m = (unsigned short*)carve((size_t)HH * FF * 2);
    unsigned short* w1m = (unsigned short*)carve((size_t)HH * HH * 2);
    unsigned short* w2m = (unsigned short*)carve((size_t)CC * HH * 2);
    unsigned short* h0b = (unsigned short*)carve((size_t)NN * HH * 2);
    float* h1f          = (float*)carve((size_t)NN * HH * 4);
    int*   idxs         = (int*)carve((size_t)NN * MAXNNZ * 4);
    float* wts          = (float*)carve((size_t)NN * MAXNNZ * 4);
    int*   cnt          = (int*)carve((size_t)NN * 4);
    float* dinv         = (float*)carve((size_t)NN * 4);
    float* agg          = (float*)carve((size_t)NN * HH * 4);
    float* colsum       = (float*)carve(HH * 4);
    float* tot          = (float*)carve(sizeof(float));
    unsigned short* h2b = (unsigned short*)carve((size_t)NN * HH * 2);
    float* out3         = (float*)carve((size_t)NN * CC * 4);

    const int nS0 = HH * FF, nS1 = HH * HH, nS2 = CC * HH;

    // 0) zero accumulators
    zero_u32<<<(65536 + 255) / 256, 256, 0, stream>>>(hist, 65536);
    zero_u32<<<(65536 + 255) / 256, 256, 0, stream>>>(hist2, 65536);
    zero_f32<<<(HH + 255) / 256, 256, 0, stream>>>(colsum, HH);
    zero_f32<<<1, 32, 0, stream>>>(tot, 1);

    // 1) exact median threshold via 2-level radix select
    hist_hi<<<(nS0 + 255) / 256, 256, 0, stream>>>(S0, nS0, hist);
    hist_hi<<<(nS1 + 255) / 256, 256, 0, stream>>>(S1, nS1, hist);
    hist_hi<<<(nS2 + 255) / 256, 256, 0, stream>>>(S2, nS2, hist);
    sel_hi<<<1, 32, 0, stream>>>(hist, SEL_K, sel);
    hist_lo<<<(nS0 + 255) / 256, 256, 0, stream>>>(S0, nS0, sel, hist2);
    hist_lo<<<(nS1 + 255) / 256, 256, 0, stream>>>(S1, nS1, sel, hist2);
    hist_lo<<<(nS2 + 255) / 256, 256, 0, stream>>>(S2, nS2, sel, hist2);
    sel_lo<<<1, 32, 0, stream>>>(hist2, sel, thr);

    // 2) masked weights -> bf16; x -> bf16
    mask_w<<<(nS0 + 255) / 256, 256, 0, stream>>>(W0, S0, thr, w0m, nS0);
    mask_w<<<(nS1 + 255) / 256, 256, 0, stream>>>(W1, S1, thr, w1m, nS1);
    mask_w<<<(nS2 + 255) / 256, 256, 0, stream>>>(W2, S2, thr, w2m, nS2);
    cvt_bf<<<((NN * FF) + 255) / 256, 256, 0, stream>>>(x, xbf, NN * FF);

    // 3) h0 = x @ W0m^T (bf16 out), h1 = h0 @ W1m^T (f32 out)
    {
        dim3 g(HH / 128, NN / 128);   // block covers 128 cols x 128 rows
        gemm_bf16<4, 2><<<g, 256, 0, stream>>>(xbf, w0m, nullptr, h0b, NN, HH, FF);
        gemm_bf16<4, 2><<<g, 256, 0, stream>>>(h0b, w1m, h1f, nullptr, NN, HH, HH);
    }

    // 4) adj -> CSR-ish (single 268 MB pass) + D^-1/2; edge weights
    scan_adj<<<NN, 256, 0, stream>>>(adj, idxs, cnt, dinv, NN);
    make_wts<<<NN, 256, 0, stream>>>(idxs, cnt, dinv, wts);

    // 5) agg = adj_n @ h1  (sparse, gathers from L2)
    spmm<<<NN, 256, 0, stream>>>(idxs, wts, cnt, h1f, agg, HH);

    // 6) pair_norm + relu -> bf16
    col_sum<<<NN / 64, 256, 0, stream>>>(agg, colsum, 64, HH);
    sq_sum<<<NN / 64, 256, 0, stream>>>(agg, colsum, tot, 64, HH, 1.0f / NN);
    pn_apply<<<((NN * HH) + 255) / 256, 256, 0, stream>>>(agg, colsum, tot, h2b,
                                                          NN * HH, HH, 1.0f / NN);

    // 7) out3 = h2 @ W2m^T  (N=64: one wave spans all 64 cols)
    {
        dim3 g(1, NN / 256);          // block covers 64 cols x 256 rows
        gemm_bf16<8, 1><<<g, 256, 0, stream>>>(h2b, w2m, out3, nullptr, NN, CC, HH);
    }

    // 8) out = adj_n @ out3
    spmm<<<NN, 64, 0, stream>>>(idxs, wts, cnt, out3, out, CC);
}
